// SlotSoftPooing_45672682226186
// MI455X (gfx1250) — compile-verified, matrix operand order checked by reading, not run on hardware
//
#include <hip/hip_runtime.h>
#include <hip/hip_bf16.h>

typedef __attribute__((ext_vector_type(16))) _Float16 v16h;
typedef __attribute__((ext_vector_type(8)))  _Float16 v8h;
typedef __attribute__((ext_vector_type(8)))  float    v8f;

#define Bz 8
#define Nz 16384
#define Dz 128
#define Kz 10
#define K16 16
#define NITER 7
#define LN_EPSF 1e-5f
#define ATT_EPSF 1e-8f
#define INV_SQRT_D 0.08838834764831845f   // 128^-0.5

// ---------------------------------------------------------------------------
// WMMA helpers.
// A-layout (documented, 16-bit 16x32): lane m = lane&15, half = lane>>4.
// Per-lane the 16 half elements are two contiguous 8-half (16B) runs:
//   K = half*8 + {0..7}  and  K = 16 + half*8 + {0..7}.
// B assumed symmetric with lane column n = lane&15 and contiguous K per lane,
// so row-major k[B,N,D], vT[B,D,N] and WT[dout][din] all stream contiguously.
// ---------------------------------------------------------------------------
__device__ __forceinline__ v16h load_frag16(const _Float16* rowp, int hf) {
  v8h lo = *(const v8h*)(rowp + hf * 8);
  v8h hi = *(const v8h*)(rowp + 16 + hf * 8);
  v16h r;
#pragma unroll
  for (int e = 0; e < 8; ++e) { r[e] = lo[e]; r[8 + e] = hi[e]; }
  return r;
}

__device__ __forceinline__ v8f wmma16(v16h a, v16h b, v8f c) {
  return __builtin_amdgcn_wmma_f32_16x16x32_f16(false, a, false, b,
                                                (short)0, c, false, false);
}

// ---------------------------------------------------------------------------
// 0) Convert Wq/Wk/Wv (f32 row-major [din][dout]) -> f16 transposed [dout][din]
// ---------------------------------------------------------------------------
__global__ void prep_weights_kernel(const float* Wq, const float* Wk, const float* Wv,
                                    _Float16* WqT, _Float16* WkT, _Float16* WvT) {
  int i = blockIdx.x * blockDim.x + threadIdx.x;
  if (i >= 3 * Dz * Dz) return;
  int mat = i / (Dz * Dz);
  int r = i % (Dz * Dz);
  int dout = r / Dz, din = r % Dz;
  const float* W = (mat == 0) ? Wq : (mat == 1) ? Wk : Wv;
  _Float16* T = (mat == 0) ? WqT : (mat == 1) ? WkT : WvT;
  T[dout * Dz + din] = (_Float16)W[din * Dz + dout];
}

// ---------------------------------------------------------------------------
// 1) slots = mu + exp(logsigma)*noise  (rows Kz..15 zero-padded)
// ---------------------------------------------------------------------------
__global__ void init_slots_kernel(const float* noise, const float* mu,
                                  const float* logsigma, float* slots) {
  int i = blockIdx.x * blockDim.x + threadIdx.x;
  if (i >= Bz * K16 * Dz) return;
  int d = i % Dz;
  int kk = (i / Dz) % K16;
  int b = i / (Dz * K16);
  float val = 0.f;
  if (kk < Kz) val = mu[d] + __expf(logsigma[d]) * noise[(b * Kz + kk) * Dz + d];
  slots[i] = val;
}

// ---------------------------------------------------------------------------
// 2) LN(x) then k = xn@Wk+bk (-> f16 [B,N,D]) and v = xn@Wv+bv (-> f16 vT[B,D,N])
//    One block = 16 rows of x; 128 threads = 4 waves; 16 WMMA d-tiles.
//    Output tiles staged in LDS, then stored branch-free as 16B vectors.
// ---------------------------------------------------------------------------
__global__ void ln_kv_kernel(const float* x, const float* g_in, const float* b_in,
                             const float* bk, const float* bv,
                             const _Float16* WkT, const _Float16* WvT,
                             _Float16* kf16, _Float16* vT) {
  __shared__ _Float16 xn[16][Dz];
  __shared__ _Float16 kout[16][Dz];
  __shared__ _Float16 vout[16][Dz];
  __shared__ float red[16][8][2];
  __shared__ float stats[16][2];
  int b = blockIdx.x / (Nz / 16);
  int n0 = (blockIdx.x % (Nz / 16)) * 16;
  int t = threadIdx.x;
  int row = t >> 3, sub = t & 7;
  const float* xr = x + ((size_t)(b * Nz + n0 + row)) * Dz;
  float s = 0.f, ss = 0.f;
#pragma unroll
  for (int i = 0; i < 16; ++i) { float v = xr[sub * 16 + i]; s += v; ss += v * v; }
  red[row][sub][0] = s; red[row][sub][1] = ss;
  __syncthreads();
  if (sub == 0) {
    float S = 0.f, SS = 0.f;
    for (int j = 0; j < 8; ++j) { S += red[row][j][0]; SS += red[row][j][1]; }
    float m = S * (1.f / Dz);
    float var = SS * (1.f / Dz) - m * m;
    stats[row][0] = m; stats[row][1] = rsqrtf(var + LN_EPSF);
  }
  __syncthreads();
  {
    float m = stats[row][0], rs = stats[row][1];
#pragma unroll
    for (int i = 0; i < 16; ++i) {
      int c = sub * 16 + i;
      xn[row][c] = (_Float16)((xr[c] - m) * rs * g_in[c] + b_in[c]);
    }
  }
  __syncthreads();
  int wave = t >> 5, lane = t & 31;
  int ln15 = lane & 15, hf = lane >> 4;
  // Hoisted A-fragments (shared by all 4 tiles this wave computes).
  v16h afr[4];
#pragma unroll
  for (int kc = 0; kc < 4; ++kc) afr[kc] = load_frag16(&xn[ln15][kc * 32], hf);
  for (int tt = wave * 4; tt < wave * 4 + 4; ++tt) {
    bool isV = (tt >= 8);
    int dt = (tt & 7) * 16;
    const _Float16* WT = isV ? WvT : WkT;
    const float* bias = isV ? bv : bk;
    v8f c = {};
#pragma unroll
    for (int kc = 0; kc < 4; ++kc) {
      v16h bb = load_frag16(WT + (dt + ln15) * Dz + kc * 32, hf);
      c = wmma16(afr[kc], bb, c);
    }
    float bsc = bias[dt + ln15];
    _Float16 (*outt)[Dz] = isV ? vout : kout;
#pragma unroll
    for (int r = 0; r < 8; ++r) {
      int orow = hf ? (r + 8) : r;
      outt[orow][dt + ln15] = (_Float16)(c[r] + bsc);
    }
  }
  __syncthreads();
  // k: row-major [B,N,D]; thread t -> (row = t>>3, 16-half chunk = t&7): 32B.
  {
    int orow = t >> 3, ch = t & 7;
    const v8h* src = (const v8h*)&kout[orow][ch * 16];
    v8h a0 = src[0], a1 = src[1];
    v8h* dst = (v8h*)(kf16 + ((size_t)(b * Nz + n0 + orow)) * Dz + ch * 16);
    dst[0] = a0; dst[1] = a1;
  }
  // vT: [B,D,N]; thread t -> column d = t, gathers 16 rows from LDS: 32B store.
  {
    int d = t;
    v8h p0, p1;
#pragma unroll
    for (int i = 0; i < 8; ++i) { p0[i] = vout[i][d]; p1[i] = vout[8 + i][d]; }
    v8h* dst = (v8h*)(vT + ((size_t)(b * Dz + d)) * Nz + n0);
    dst[0] = p0; dst[1] = p1;
  }
}

// ---------------------------------------------------------------------------
// 3) q = LN(slots)@Wq + bq  -> f16 [B,16,D].  One block per batch (tiny).
// ---------------------------------------------------------------------------
__global__ void qproj_kernel(const float* slots, const float* g_sl, const float* b_sl,
                             const float* bq, const _Float16* WqT, _Float16* qf16) {
  __shared__ _Float16 lnr[16][Dz];
  __shared__ _Float16 qout[16][Dz];
  __shared__ float red[16][8][2];
  __shared__ float stats[16][2];
  int b = blockIdx.x;
  int t = threadIdx.x;
  int row = t >> 3, sub = t & 7;
  const float* sr = slots + (b * K16 + row) * Dz;
  float s = 0.f, ss = 0.f;
#pragma unroll
  for (int i = 0; i < 16; ++i) { float v = sr[sub * 16 + i]; s += v; ss += v * v; }
  red[row][sub][0] = s; red[row][sub][1] = ss;
  __syncthreads();
  if (sub == 0) {
    float S = 0.f, SS = 0.f;
    for (int j = 0; j < 8; ++j) { S += red[row][j][0]; SS += red[row][j][1]; }
    float m = S * (1.f / Dz);
    float var = SS * (1.f / Dz) - m * m;
    stats[row][0] = m; stats[row][1] = rsqrtf(var + LN_EPSF);
  }
  __syncthreads();
  {
    float m = stats[row][0], rs = stats[row][1];
#pragma unroll
    for (int i = 0; i < 16; ++i) {
      int c = sub * 16 + i;
      lnr[row][c] = (_Float16)((sr[c] - m) * rs * g_sl[c] + b_sl[c]);
    }
  }
  __syncthreads();
  int wave = t >> 5, lane = t & 31;
  int ln15 = lane & 15, hf = lane >> 4;
  v16h afr[4];
#pragma unroll
  for (int kc = 0; kc < 4; ++kc) afr[kc] = load_frag16(&lnr[ln15][kc * 32], hf);
  for (int tt = wave * 2; tt < wave * 2 + 2; ++tt) {
    int dt = tt * 16;
    v8f c = {};
#pragma unroll
    for (int kc = 0; kc < 4; ++kc) {
      v16h bb = load_frag16(WqT + (dt + ln15) * Dz + kc * 32, hf);
      c = wmma16(afr[kc], bb, c);
    }
    float bsc = bq[dt + ln15];
#pragma unroll
    for (int r = 0; r < 8; ++r) {
      int orow = hf ? (r + 8) : r;
      qout[orow][dt + ln15] = (_Float16)(c[r] + bsc);
    }
  }
  __syncthreads();
  {
    int orow = t >> 3, ch = t & 7;
    const v8h* src = (const v8h*)&qout[orow][ch * 16];
    v8h a0 = src[0], a1 = src[1];
    v8h* dst = (v8h*)(qf16 + (b * K16 + orow) * Dz + ch * 16);
    dst[0] = a0; dst[1] = a1;
  }
}

// ---------------------------------------------------------------------------
// 4) dots[b,k,n] = scale * q[b,k,:]·k[b,n,:].  1 wave => one 16x16 tile.
//    Tile staged in LDS -> two b128 stores per lane.
// ---------------------------------------------------------------------------
__global__ void dots_kernel(const _Float16* qf16, const _Float16* kf16, float* dots) {
  __shared__ float dtile[4][16][16];
  int b = blockIdx.x / (Nz / 64);
  int nblk = blockIdx.x % (Nz / 64);
  int wave = threadIdx.x >> 5, lane = threadIdx.x & 31;
  int n0w = nblk * 64 + wave * 16;
  int ln15 = lane & 15, hf = lane >> 4;
  const _Float16* qrow = qf16 + (b * K16 + ln15) * Dz;
  const _Float16* krow = kf16 + ((size_t)(b * Nz + n0w + ln15)) * Dz;
  v8f c = {};
#pragma unroll
  for (int kc = 0; kc < 4; ++kc) {
    v16h a = load_frag16(qrow + kc * 32, hf);
    v16h bb = load_frag16(krow + kc * 32, hf);
    c = wmma16(a, bb, c);
  }
#pragma unroll
  for (int r = 0; r < 8; ++r)
    dtile[wave][hf ? (r + 8) : r][ln15] = c[r] * INV_SQRT_D;
  __syncthreads();
  // lane -> (row = lane>>1, half-row = lane&1): 8 floats = 32B contiguous.
  int orow = lane >> 1, hrow = lane & 1;
  const float4* src = (const float4*)&dtile[wave][orow][hrow * 8];
  float4 v0 = src[0], v1 = src[1];
  float4* dst = (float4*)(dots + ((size_t)b * K16 + orow) * Nz + n0w + hrow * 8);
  dst[0] = v0; dst[1] = v1;
}

// ---------------------------------------------------------------------------
// 5) softmax over slots (K axis) per (b,n); write attn f32 and (attn+eps) f16.
//    Also emit deterministic per-block partial sums of (attn+eps) over n.
// ---------------------------------------------------------------------------
__global__ void softmax_kernel(const float* dots, float* attn, _Float16* attn_e,
                               float* psum) {
  __shared__ float sdata[256];
  int b = blockIdx.x / (Nz / 256);
  int blk = blockIdx.x % (Nz / 256);
  int n = blk * 256 + threadIdx.x;
  float vals[Kz];
  float mx = -1e30f;
#pragma unroll
  for (int k = 0; k < Kz; ++k) {
    float v = dots[((size_t)b * K16 + k) * Nz + n];
    vals[k] = v; mx = fmaxf(mx, v);
  }
  float s = 0.f;
#pragma unroll
  for (int k = 0; k < Kz; ++k) { vals[k] = __expf(vals[k] - mx); s += vals[k]; }
  float inv = 1.f / s;
  float ev[Kz];
#pragma unroll
  for (int k = 0; k < Kz; ++k) {
    float a = vals[k] * inv;
    attn[((size_t)b * K16 + k) * Nz + n] = a;
    ev[k] = a + ATT_EPSF;
    attn_e[((size_t)b * K16 + k) * Nz + n] = (_Float16)ev[k];
  }
#pragma unroll
  for (int k = Kz; k < K16; ++k)
    attn_e[((size_t)b * K16 + k) * Nz + n] = (_Float16)0.f;
  for (int k = 0; k < Kz; ++k) {
    sdata[threadIdx.x] = ev[k];
    __syncthreads();
    for (int off = 128; off > 0; off >>= 1) {
      if (threadIdx.x < off) sdata[threadIdx.x] += sdata[threadIdx.x + off];
      __syncthreads();
    }
    if (threadIdx.x == 0) psum[(b * K16 + k) * (Nz / 256) + blk] = sdata[0];
    __syncthreads();
  }
}

// 5b) reduce partial sums -> asum[b*16+k]  (deterministic, no atomics)
__global__ void asum_reduce_kernel(const float* psum, float* asum) {
  int t = threadIdx.x;           // 128 = B*K16
  int k = t % K16;
  float s = 1.f;                 // benign value for padded rows
  if (k < Kz) {
    s = 0.f;
    for (int j = 0; j < Nz / 256; ++j) s += psum[t * (Nz / 256) + j];
  }
  asum[t] = s;
}

// ---------------------------------------------------------------------------
// 6) updates[b,k,d] = (1/asum) * sum_n attn_e[b,k,n] * v[b,n,d]  via WMMA.
//    Block = (b, d-tile); 4 waves split N, LDS cross-wave reduce.
// ---------------------------------------------------------------------------
__global__ void updates_kernel(const _Float16* attn_e, const _Float16* vT,
                               const float* asum, float* upd) {
  __shared__ float acc[4][8][32];
  int b = blockIdx.x >> 3;
  int dt = (blockIdx.x & 7) * 16;
  int wave = threadIdx.x >> 5, lane = threadIdx.x & 31;
  int ln15 = lane & 15, hf = lane >> 4;
  const _Float16* arow = attn_e + ((size_t)b * K16 + ln15) * Nz;
  const _Float16* vrow = vT + ((size_t)(b * Dz + dt + ln15)) * Nz;
  v8f c = {};
  for (int nc = wave; nc < Nz / 32; nc += 4) {
    int nb = nc * 32;
    __builtin_prefetch(vrow + nb + 512, 0, 1);   // -> global_prefetch_b8
    __builtin_prefetch(arow + nb + 512, 0, 1);
    v16h a = load_frag16(arow + nb, hf);
    v16h bb = load_frag16(vrow + nb, hf);
    c = wmma16(a, bb, c);
  }
#pragma unroll
  for (int r = 0; r < 8; ++r) acc[wave][r][lane] = c[r];
  __syncthreads();
  if (wave == 0) {
#pragma unroll
    for (int r = 0; r < 8; ++r) {
      float v = acc[0][r][lane] + acc[1][r][lane] + acc[2][r][lane] + acc[3][r][lane];
      int m = hf ? (r + 8) : r;
      float res = (m < Kz) ? (v / asum[b * K16 + m]) : 0.f;
      upd[(b * K16 + m) * Dz + dt + ln15] = res;
    }
  }
}

// ---------------------------------------------------------------------------
// 7) GRU cell + LN + MLP residual over one (b, slot) row. 128 threads/row.
// ---------------------------------------------------------------------------
__global__ void gru_mlp_kernel(const float* upd, float* slots,
                               const float* Wih, const float* Whh,
                               const float* bih, const float* bhh,
                               const float* W1, const float* b1,
                               const float* W2, const float* b2,
                               const float* g_ff, const float* b_ff) {
  __shared__ float u[Dz], p[Dz], h[Dz], lnr[Dz], hid[Dz];
  __shared__ float red[2][Dz];
  int b = blockIdx.x / Kz, k = blockIdx.x % Kz;
  int t = threadIdx.x;
  u[t] = upd[(b * K16 + k) * Dz + t];
  p[t] = slots[(b * K16 + k) * Dz + t];
  __syncthreads();
  float ir = bih[t], iz = bih[t + Dz], inn = bih[t + 2 * Dz];
  float hr = bhh[t], hz = bhh[t + Dz], hn = bhh[t + 2 * Dz];
  const float* wi0 = Wih + t * Dz;
  const float* wi1 = Wih + (t + Dz) * Dz;
  const float* wi2 = Wih + (t + 2 * Dz) * Dz;
  const float* wh0 = Whh + t * Dz;
  const float* wh1 = Whh + (t + Dz) * Dz;
  const float* wh2 = Whh + (t + 2 * Dz) * Dz;
  for (int d = 0; d < Dz; ++d) {
    float ud = u[d], pd = p[d];
    ir += wi0[d] * ud; iz += wi1[d] * ud; inn += wi2[d] * ud;
    hr += wh0[d] * pd; hz += wh1[d] * pd; hn += wh2[d] * pd;
  }
  float r = 1.f / (1.f + __expf(-(ir + hr)));
  float z = 1.f / (1.f + __expf(-(iz + hz)));
  float nn = tanhf(inn + r * hn);
  float hnew = (1.f - z) * nn + z * p[t];
  h[t] = hnew;
  red[0][t] = hnew; red[1][t] = hnew * hnew;
  __syncthreads();
  for (int off = 64; off > 0; off >>= 1) {
    if (t < off) { red[0][t] += red[0][t + off]; red[1][t] += red[1][t + off]; }
    __syncthreads();
  }
  float m = red[0][0] * (1.f / Dz);
  float var = red[1][0] * (1.f / Dz) - m * m;
  float rs = rsqrtf(var + LN_EPSF);
  lnr[t] = (h[t] - m) * rs * g_ff[t] + b_ff[t];
  __syncthreads();
  float a1 = b1[t];
  for (int d = 0; d < Dz; ++d) a1 += lnr[d] * W1[d * Dz + t];
  hid[t] = fmaxf(a1, 0.f);
  __syncthreads();
  float a2 = b2[t];
  for (int d = 0; d < Dz; ++d) a2 += hid[d] * W2[d * Dz + t];
  slots[(b * K16 + k) * Dz + t] = h[t] + a2;
}

// ---------------------------------------------------------------------------
// 8) Outputs: node features, ones-adjacency, transposed attention.
// ---------------------------------------------------------------------------
__global__ void out_nodes_kernel(const float* slots, float* out) {
  int i = blockIdx.x * blockDim.x + threadIdx.x;
  if (i >= Bz * Kz * Dz) return;
  int d = i % Dz;
  int k = (i / Dz) % Kz;
  int b = i / (Dz * Kz);
  out[i] = slots[(b * K16 + k) * Dz + d];
}

__global__ void out_adj_kernel(float* out) {
  int i = blockIdx.x * blockDim.x + threadIdx.x;
  if (i < Bz * Kz * Kz) out[i] = 1.f;
}

__global__ void out_s_kernel(const float* attn, float* out) {
  int i = blockIdx.x * blockDim.x + threadIdx.x;
  if (i >= Bz * Nz) return;
  int n = i % Nz;
  int b = i / Nz;
  float* o = out + ((size_t)(b * Nz + n)) * Kz;
#pragma unroll
  for (int k = 0; k < Kz; ++k) o[k] = attn[((size_t)b * K16 + k) * Nz + n];
}

// ---------------------------------------------------------------------------
extern "C" void kernel_launch(void* const* d_in, const int* in_sizes, int n_in,
                              void* d_out, int out_size, void* d_ws, size_t ws_size,
                              hipStream_t stream) {
  (void)in_sizes; (void)n_in; (void)out_size; (void)ws_size;
  const float* x        = (const float*)d_in[0];
  const float* noise    = (const float*)d_in[1];
  const float* mu       = (const float*)d_in[2];
  const float* logsigma = (const float*)d_in[3];
  const float* Wq  = (const float*)d_in[4];   const float* bq  = (const float*)d_in[5];
  const float* Wk  = (const float*)d_in[6];   const float* bk  = (const float*)d_in[7];
  const float* Wv  = (const float*)d_in[8];   const float* bv  = (const float*)d_in[9];
  const float* Wih = (const float*)d_in[10];  const float* Whh = (const float*)d_in[11];
  const float* bih = (const float*)d_in[12];  const float* bhh = (const float*)d_in[13];
  const float* W1  = (const float*)d_in[14];  const float* b1  = (const float*)d_in[15];
  const float* W2  = (const float*)d_in[16];  const float* b2  = (const float*)d_in[17];
  const float* g_in = (const float*)d_in[18]; const float* b_in = (const float*)d_in[19];
  const float* g_sl = (const float*)d_in[20]; const float* b_sl = (const float*)d_in[21];
  const float* g_ff = (const float*)d_in[22]; const float* b_ff = (const float*)d_in[23];

  char* wsb = (char*)d_ws;
  size_t off = 0;
  auto take = [&](size_t bytes) -> char* {
    char* p = wsb + off;
    off = (off + bytes + 255) & ~(size_t)255;
    return p;
  };
  _Float16* kf16   = (_Float16*)take((size_t)Bz * Nz * Dz * 2);
  _Float16* vT     = (_Float16*)take((size_t)Bz * Nz * Dz * 2);
  _Float16* WqT    = (_Float16*)take((size_t)Dz * Dz * 2);
  _Float16* WkT    = (_Float16*)take((size_t)Dz * Dz * 2);
  _Float16* WvT    = (_Float16*)take((size_t)Dz * Dz * 2);
  float*    slots  = (float*)take((size_t)Bz * K16 * Dz * 4);
  _Float16* qf16   = (_Float16*)take((size_t)Bz * K16 * Dz * 2);
  float*    dots   = (float*)take((size_t)Bz * K16 * Nz * 4);
  float*    attn   = (float*)take((size_t)Bz * K16 * Nz * 4);
  _Float16* attn_e = (_Float16*)take((size_t)Bz * K16 * Nz * 2);
  float*    psum   = (float*)take((size_t)Bz * K16 * (Nz / 256) * 4);
  float*    asum   = (float*)take((size_t)Bz * K16 * 4);
  float*    upd    = (float*)take((size_t)Bz * K16 * Dz * 4);

  prep_weights_kernel<<<(3 * Dz * Dz + 255) / 256, 256, 0, stream>>>(
      Wq, Wk, Wv, WqT, WkT, WvT);
  init_slots_kernel<<<(Bz * K16 * Dz + 255) / 256, 256, 0, stream>>>(
      noise, mu, logsigma, slots);
  ln_kv_kernel<<<Bz * (Nz / 16), 128, 0, stream>>>(
      x, g_in, b_in, bk, bv, WkT, WvT, kf16, vT);

  for (int it = 0; it < NITER; ++it) {
    qproj_kernel<<<Bz, 128, 0, stream>>>(slots, g_sl, b_sl, bq, WqT, qf16);
    dots_kernel<<<Bz * (Nz / 64), 128, 0, stream>>>(qf16, kf16, dots);
    softmax_kernel<<<Bz * (Nz / 256), 256, 0, stream>>>(dots, attn, attn_e, psum);
    asum_reduce_kernel<<<1, 128, 0, stream>>>(psum, asum);
    updates_kernel<<<Bz * 8, 128, 0, stream>>>(attn_e, vT, asum, upd);
    gru_mlp_kernel<<<Bz * Kz, 128, 0, stream>>>(
        upd, slots, Wih, Whh, bih, bhh, W1, b1, W2, b2, g_ff, b_ff);
  }

  float* out = (float*)d_out;
  out_nodes_kernel<<<(Bz * Kz * Dz + 255) / 256, 256, 0, stream>>>(slots, out);
  out_adj_kernel<<<(Bz * Kz * Kz + 255) / 256, 256, 0, stream>>>(out + Bz * Kz * Dz);
  out_s_kernel<<<(Bz * Nz + 255) / 256, 256, 0, stream>>>(
      attn, out + Bz * Kz * Dz + Bz * Kz * Kz);
}
